// SelfAttention_48868137894540
// MI455X (gfx1250) — matrix operand order repaired and checked by reading, not compiled, not run on hardware
//
#include <hip/hip_runtime.h>
#include <hip/hip_bf16.h>

// ---------------------------------------------------------------------------
// Problem constants (B,S,D,H from the reference)
// ---------------------------------------------------------------------------
constexpr int kB  = 4;
constexpr int kS  = 2048;
constexpr int kD  = 512;
constexpr int kH  = 8;
constexpr int kHD = kH * kD;     // 4096
constexpr float kScale = 0.044194173824159216f; // 1/sqrt(512)

typedef __attribute__((ext_vector_type(16))) __bf16 v16bf;
typedef __attribute__((ext_vector_type(8)))  __bf16 v8bf;
typedef __attribute__((ext_vector_type(4)))  __bf16 v4bf;
typedef __attribute__((ext_vector_type(8)))  float  v8f;
typedef __attribute__((ext_vector_type(4)))  unsigned int v4u;
typedef __attribute__((ext_vector_type(8)))  int v8i;
typedef __attribute__((ext_vector_type(4)))  int v4i;

// ---------------------------------------------------------------------------
// gfx1250 async copy global->LDS (ASYNCcnt-tracked), inline asm.
// LDS byte address = low 32 bits of generic pointer (ISA aperture rule).
// ---------------------------------------------------------------------------
__device__ __forceinline__ void async_copy_b128(void* lds, const void* g) {
    asm volatile("global_load_async_to_lds_b128 %0, %1, off"
                 :: "v"((unsigned)(size_t)lds),
                    "v"((unsigned long long)(size_t)g)
                 : "memory");
}
__device__ __forceinline__ void wait_async0() {
    asm volatile("s_wait_asynccnt 0x0" ::: "memory");
}
__device__ __forceinline__ void wait_tensor0() {
#if __has_builtin(__builtin_amdgcn_s_wait_tensorcnt)
    __builtin_amdgcn_s_wait_tensorcnt(0);
#else
    asm volatile("s_wait_tensorcnt 0x0" ::: "memory");
#endif
}

// ---------------------------------------------------------------------------
// fp32 -> bf16 conversion (4-wide)
// ---------------------------------------------------------------------------
__global__ void cvt_f32_bf16(const float* __restrict__ src,
                             __bf16* __restrict__ dst, int n) {
    int i = (blockIdx.x * blockDim.x + threadIdx.x) * 4;
    if (i < n) {
        float4 v = *reinterpret_cast<const float4*>(src + i);
        v4bf o;
        o[0] = (__bf16)v.x; o[1] = (__bf16)v.y;
        o[2] = (__bf16)v.z; o[3] = (__bf16)v.w;
        *reinterpret_cast<v4bf*>(dst + i) = o;
    }
}

// ---------------------------------------------------------------------------
// Generic NT bf16 GEMM: C[M,N] = A[M,K] * B[N,K]^T  (both row-major)
// Double-buffered LDS tiles filled by async-to-LDS copies; B fragments
// prefetched into registers so the 8 WMMAs issue back-to-back.
// ---------------------------------------------------------------------------
#define GEMM_LDT 40   // 32 + 8 bf16 padding

__global__ __launch_bounds__(256)
void gemm_nt_bf16(const __bf16* __restrict__ A, const __bf16* __restrict__ Bw,
                  void* __restrict__ Cout,
                  long long strideAb, long long strideBh, long long strideC,
                  int Hh, int M, int N, int K, int ldc, int c_is_f32) {
    __shared__ __bf16 As[2][128 * GEMM_LDT];
    __shared__ __bf16 Bs[2][128 * GEMM_LDT];

    const int z  = blockIdx.z;
    const int bb = z / Hh, hh = z % Hh;
    const __bf16* Ap = A  + (long long)bb * strideAb;
    const __bf16* Bp = Bw + (long long)hh * strideBh;

    const int row0 = blockIdx.x * 128;
    const int col0 = blockIdx.y * 128;
    const int tid  = threadIdx.x;
    const int wave = tid >> 5;
    const int lane = tid & 31;
    const int nl   = lane & 15;        // N (B/C frags) / M (A frag) lane index
    const int hi   = lane >> 4;        // lane group
    const int k0a  = hi ? 8 : 0;       // A-frag K offset  (ISA 16-bit A layout)
    const int k0b  = hi ? 16 : 0;      // unused; B frags use k0a (same k order as A)
    const int mbase = hi * 8;          // C-frag row base

    v8f acc[8];
#pragma unroll
    for (int i = 0; i < 8; i++)
#pragma unroll
        for (int r = 0; r < 8; r++) acc[i][r] = 0.0f;

    const int cpr = tid >> 1;          // staging row 0..127
    const int cpc = (tid & 1) * 16;    // staging 16-elem column half

    auto stage_async = [&](int kb, int buf) {
        __bf16* la = &As[buf][cpr * GEMM_LDT + cpc];
        const __bf16* ga = Ap + (long long)(row0 + cpr) * K + kb + cpc;
        async_copy_b128(la, ga);
        async_copy_b128(la + 8, ga + 8);
        __bf16* lb = &Bs[buf][cpr * GEMM_LDT + cpc];
        const __bf16* gb = Bp + (long long)(col0 + cpr) * K + kb + cpc;
        async_copy_b128(lb, gb);
        async_copy_b128(lb + 8, gb + 8);
    };

    const int nk = K / 32;
    stage_async(0, 0);
    wait_async0();
    __syncthreads();

    for (int kc = 0; kc < nk; kc++) {
        const int buf = kc & 1;
        if (kc + 1 < nk) stage_async((kc + 1) * 32, buf ^ 1);

        // A fragment for this wave's 16-row strip
        v16bf af;
        {
            const __bf16* p = &As[buf][(wave * 16 + nl) * GEMM_LDT];
            v8bf lo = *reinterpret_cast<const v8bf*>(p + k0a);
            v8bf hh2 = *reinterpret_cast<const v8bf*>(p + k0a + 16);
#pragma unroll
            for (int i = 0; i < 8; i++) { af[i] = lo[i]; af[i + 8] = hh2[i]; }
        }
        // prefetch all 8 B fragments, then burst the WMMAs
        v16bf bfr[8];
#pragma unroll
        for (int nc = 0; nc < 8; nc++) {
            const __bf16* p = &Bs[buf][(nc * 16 + nl) * GEMM_LDT + k0a];
            v8bf b0 = *reinterpret_cast<const v8bf*>(p);
            v8bf b1 = *reinterpret_cast<const v8bf*>(p + 16);
#pragma unroll
            for (int i = 0; i < 8; i++) { bfr[nc][i] = b0[i]; bfr[nc][i + 8] = b1[i]; }
        }
#pragma unroll
        for (int nc = 0; nc < 8; nc++)
            acc[nc] = __builtin_amdgcn_wmma_f32_16x16x32_bf16(
                false, af, false, bfr[nc], (short)0, acc[nc], false, false);

        wait_async0();
        __syncthreads();
    }

    // ---- store C (fp32 or bf16) ----
#pragma unroll
    for (int nc = 0; nc < 8; nc++) {
        const int col = col0 + nc * 16 + nl;
        if (c_is_f32) {
            float* C = reinterpret_cast<float*>(Cout) + (long long)z * strideC;
#pragma unroll
            for (int r = 0; r < 8; r++)
                C[(long long)(row0 + wave * 16 + mbase + r) * ldc + col] = acc[nc][r];
        } else {
            __bf16* C = reinterpret_cast<__bf16*>(Cout) + (long long)z * strideC;
#pragma unroll
            for (int r = 0; r < 8; r++)
                C[(long long)(row0 + wave * 16 + mbase + r) * ldc + col] =
                    (__bf16)acc[nc][r];
        }
    }
}

// ---------------------------------------------------------------------------
// Fused flash attention. K tile staged by the Tensor Data Mover (wave 0
// issues one tensor_load_to_lds with LDS row padding; TENSORcnt-waited),
// V tile staged transposed by all threads in parallel with the TDM copy.
// ---------------------------------------------------------------------------
#define LDK 520   // 512 + 8 pad (K tile row stride, bf16)
#define LDV 40    // 32 + 8 pad (V^T row stride, bf16)
#define LDP 40    // P tile row stride

__global__ __launch_bounds__(256)
void flash_attn_bf16(const __bf16* __restrict__ Qb, const __bf16* __restrict__ Kb,
                     const __bf16* __restrict__ Vb, const int* __restrict__ mask,
                     __bf16* __restrict__ cat) {
    extern __shared__ char smem[];
    __bf16* Ks = reinterpret_cast<__bf16*>(smem);           // 32 x LDK
    __bf16* Vt = Ks + 32 * LDK;                             // 512 x LDV (V^T)
    float*  Ls = reinterpret_cast<float*>(Vt + 512 * LDV);  // 64 x 32 logits
    __bf16* Ps = reinterpret_cast<__bf16*>(Ls + 64 * 32);   // 64 x LDP probs
    float*  mst = reinterpret_cast<float*>(Ps + 64 * LDP);  // running max [64]
    float*  lst = mst + 64;                                 // running sum [64]
    float*  ast = lst + 64;                                 // alpha [64]
    int*    qm  = reinterpret_cast<int*>(ast + 64);         // query mask [64]
    int*    km  = qm + 64;                                  // key mask [32]

    const int bh   = blockIdx.x;
    const int sblk = blockIdx.y;
    const int b = bh / kH, h = bh % kH;
    const int tid = threadIdx.x, wave = tid >> 5, lane = tid & 31;
    const int rg = wave >> 1;          // row group 0..3
    const int ch = wave & 1;           // column half of D
    const int nl = lane & 15;
    const int hi = lane >> 4;
    const int k0a = hi ? 8 : 0, k0b = hi ? 16 : 0, mbase = hi * 8;
    const int s0 = sblk * 64;

    if (tid < 64) {
        mst[tid] = -1e30f;
        lst[tid] = 0.0f;
        qm[tid]  = mask[b * kS + s0 + tid];
    }

    // ---- resident Q fragments: rows rg*16..+15, all 16 K-chunks ----
    const __bf16* qrow = Qb + ((long long)bh * kS + s0 + rg * 16 + nl) * kD;
    v16bf qf[16];
#pragma unroll
    for (int kc = 0; kc < 16; kc++) {
        v8bf lo = *reinterpret_cast<const v8bf*>(qrow + kc * 32 + k0a);
        v8bf hh2 = *reinterpret_cast<const v8bf*>(qrow + kc * 32 + k0a + 16);
#pragma unroll
        for (int i = 0; i < 8; i++) { qf[kc][i] = lo[i]; qf[kc][i + 8] = hh2[i]; }
    }

    v8f o[16];
#pragma unroll
    for (int i = 0; i < 16; i++)
#pragma unroll
        for (int r = 0; r < 8; r++) o[i][r] = 0.0f;

    for (int t0 = 0; t0 < kS; t0 += 32) {
        // ---- TDM: stage K tile 32x512 bf16 into Ks with +16B/row padding ----
        const __bf16* ktile = Kb + ((long long)bh * kS + t0) * kD;
#if __has_builtin(__builtin_amdgcn_tensor_load_to_lds)
        if (wave == 0) {
            unsigned long long ga = (unsigned long long)(size_t)ktile;
            v4u g0;
            g0[0] = 1u;                                  // count=1 (user D#)
            g0[1] = (unsigned)(size_t)Ks;                // lds_addr
            g0[2] = (unsigned)ga;                        // global_addr[31:0]
            g0[3] = (unsigned)(ga >> 32) | 0x80000000u;  // addr[56:32] | type=2
            v8i g1;
            g1[0] = (int)((1u << 16)      // data_size = 2 bytes
                        | (1u << 20)      // pad_enable
                        | (7u << 22)      // pad_interval = 256 DWORDs (one row)
                        | (3u << 25));    // pad_amount   = 4 DWORDs (16 bytes)
            g1[1] = (int)(512u << 16);    // tensor_dim0 = 512 (bits 79:48)
            g1[2] = (int)(2048u << 16);   // tensor_dim1 = 2048 (bits 111:80)
            g1[3] = (int)(512u << 16);    // tile_dim0 = 512 (bits 127:112)
            g1[4] = 32;                   // tile_dim1 = 32, tile_dim2 = 0
            g1[5] = 512;                  // tensor_dim0_stride = 512
            g1[6] = 0;
            g1[7] = 0;
            v4i z4 = {0, 0, 0, 0};
            v8i z8 = {0, 0, 0, 0, 0, 0, 0, 0};
            __builtin_amdgcn_tensor_load_to_lds(g0, g1, z4, z4, z8, 0);
        }
#else
        {
            const int tr = tid >> 3;
            const int c0 = (tid & 7) * 64;
            __bf16* s = Ks + tr * LDK + c0;
            const __bf16* gk = ktile + (long long)tr * kD + c0;
#pragma unroll
            for (int j = 0; j < 8; j++)
                async_copy_b128(s + j * 8, gk + j * 8);
        }
#endif
        // ---- stage V tile transposed (overlaps with TDM copy) ----
        {
            const int tv = tid & 31;
            const int cv = (tid >> 5) * 64;
            const __bf16* gv = Vb + ((long long)bh * kS + t0 + tv) * kD + cv;
#pragma unroll
            for (int jj = 0; jj < 8; jj++) {
                v8bf v = reinterpret_cast<const v8bf*>(gv)[jj];
#pragma unroll
                for (int e = 0; e < 8; e++)
                    Vt[(cv + jj * 8 + e) * LDV + tv] = v[e];
            }
        }
        if (tid < 32) km[tid] = mask[b * kS + t0 + tid];
#if __has_builtin(__builtin_amdgcn_tensor_load_to_lds)
        if (wave == 0) wait_tensor0();
#else
        wait_async0();
#endif
        __syncthreads();

        // ---- logits tile: 16 rows (rg) x 16 keys (ch half), pipelined ----
        auto loadKB = [&](int kc) {
            const __bf16* p = Ks + (ch * 16 + nl) * LDK + kc * 32 + k0a;
            v8bf b0 = *reinterpret_cast<const v8bf*>(p);
            v8bf b1 = *reinterpret_cast<const v8bf*>(p + 16);
            v16bf bf;
#pragma unroll
            for (int i = 0; i < 8; i++) { bf[i] = b0[i]; bf[i + 8] = b1[i]; }
            return bf;
        };
        v8f lg;
#pragma unroll
        for (int r = 0; r < 8; r++) lg[r] = 0.0f;
        v16bf bc = loadKB(0);
#pragma unroll
        for (int kc = 0; kc < 16; kc++) {
            v16bf bn = (kc < 15) ? loadKB(kc + 1) : bc;
            lg = __builtin_amdgcn_wmma_f32_16x16x32_bf16(
                false, qf[kc], false, bc, (short)0, lg, false, false);
            bc = bn;
        }
#pragma unroll
        for (int r = 0; r < 8; r++)
            Ls[(rg * 16 + mbase + r) * 32 + ch * 16 + nl] = lg[r] * kScale;
        __syncthreads();

        // ---- online softmax: one thread per query row ----
        if (tid < 64) {
            const int q = qm[tid];
            float mo = mst[tid];
            float rowmax = -1e30f;
            float lv[32];
#pragma unroll
            for (int j = 0; j < 32; j++) {
                float v = Ls[tid * 32 + j];
                if (!(q && km[j])) v = -1e9f;
                lv[j] = v;
                rowmax = fmaxf(rowmax, v);
            }
            const float mn = fmaxf(mo, rowmax);
            const float al = __expf(mo - mn);
            float sum = 0.0f;
#pragma unroll
            for (int j = 0; j < 32; j++) {
                float p = __expf(lv[j] - mn);
                sum += p;
                Ps[tid * LDP + j] = (__bf16)p;
            }
            mst[tid] = mn;
            lst[tid] = lst[tid] * al + sum;
            ast[tid] = al;
        }
        __syncthreads();

        // ---- rescale O, then O += P @ V (pipelined V fragments) ----
        float av[8];
#pragma unroll
        for (int r = 0; r < 8; r++) av[r] = ast[rg * 16 + mbase + r];
#pragma unroll
        for (int nc = 0; nc < 16; nc++)
#pragma unroll
            for (int r = 0; r < 8; r++) o[nc][r] *= av[r];

        v16bf pf;
        {
            const __bf16* p = Ps + (rg * 16 + nl) * LDP;
            v8bf lo3 = *reinterpret_cast<const v8bf*>(p + k0a);
            v8bf h3  = *reinterpret_cast<const v8bf*>(p + k0a + 16);
#pragma unroll
            for (int i = 0; i < 8; i++) { pf[i] = lo3[i]; pf[i + 8] = h3[i]; }
        }
        auto loadVB = [&](int nc) {
            const __bf16* p = Vt + (ch * 256 + nc * 16 + nl) * LDV + k0a;
            v8bf b0 = *reinterpret_cast<const v8bf*>(p);
            v8bf b1 = *reinterpret_cast<const v8bf*>(p + 16);
            v16bf bf;
#pragma unroll
            for (int i = 0; i < 8; i++) { bf[i] = b0[i]; bf[i + 8] = b1[i]; }
            return bf;
        };
        v16bf vc = loadVB(0);
#pragma unroll
        for (int nc = 0; nc < 16; nc++) {
            v16bf vn = (nc < 15) ? loadVB(nc + 1) : vc;
            o[nc] = __builtin_amdgcn_wmma_f32_16x16x32_bf16(
                false, pf, false, vc, (short)0, o[nc], false, false);
            vc = vn;
        }
        __syncthreads();
    }

    // ---- normalize by l and store into concat-heads layout [B,S,H*D] ----
    float li[8];
#pragma unroll
    for (int r = 0; r < 8; r++) {
        float l = lst[rg * 16 + mbase + r];
        li[r] = (l > 0.0f) ? (1.0f / l) : 0.0f;
    }
#pragma unroll
    for (int nc = 0; nc < 16; nc++) {
#pragma unroll
        for (int r = 0; r < 8; r++) {
            const int srow = s0 + rg * 16 + mbase + r;
            const int col  = h * kD + ch * 256 + nc * 16 + nl;
            cat[((long long)b * kS + srow) * kHD + col] = (__bf16)(o[nc][r] * li[r]);
        }
    }
}

// ---------------------------------------------------------------------------
// Host side
// ---------------------------------------------------------------------------
extern "C" void kernel_launch(void* const* d_in, const int* in_sizes, int n_in,
                              void* d_out, int out_size, void* d_ws, size_t ws_size,
                              hipStream_t stream) {
    const float* x    = (const float*)d_in[0];
    const int*   mask = (const int*)  d_in[1];
    const float* Wq   = (const float*)d_in[2];
    const float* Wk   = (const float*)d_in[3];
    const float* Wv   = (const float*)d_in[4];
    const float* Wp   = (const float*)d_in[5];
    float* out = (float*)d_out;

    char* ws = (char*)d_ws;
    const long long nX  = (long long)kB * kS * kD;
    const long long nW  = (long long)kH * kD * kD;
    const long long nWp = (long long)kD * kHD;
    const long long nQ  = (long long)kB * kH * kS * kD;
    const long long nCat = (long long)kB * kS * kHD;

    size_t off = 0;
    __bf16* xb  = (__bf16*)(ws + off); off += (size_t)nX  * 2;
    __bf16* Wqb = (__bf16*)(ws + off); off += (size_t)nW  * 2;
    __bf16* Wkb = (__bf16*)(ws + off); off += (size_t)nW  * 2;
    __bf16* Wvb = (__bf16*)(ws + off); off += (size_t)nW  * 2;
    __bf16* Wpb = (__bf16*)(ws + off); off += (size_t)nWp * 2;
    __bf16* Qb  = (__bf16*)(ws + off); off += (size_t)nQ  * 2;
    __bf16* Kbf = (__bf16*)(ws + off); off += (size_t)nQ  * 2;
    __bf16* Vbf = (__bf16*)(ws + off); off += (size_t)nQ  * 2;
    __bf16* cat = (__bf16*)(ws + off); off += (size_t)nCat * 2;

    auto cvt = [&](const float* s, __bf16* d, long long n) {
        int blocks = (int)((n / 4 + 255) / 256);
        cvt_f32_bf16<<<blocks, 256, 0, stream>>>(s, d, (int)n);
    };
    cvt(x,  xb,  nX);
    cvt(Wq, Wqb, nW);
    cvt(Wk, Wkb, nW);
    cvt(Wv, Wvb, nW);
    cvt(Wp, Wpb, nWp);

    // Q/K/V projections: C[z = S x D] = x[b] (SxD) * W[h]^T, z = b*H + h
    dim3 gproj(kS / 128, kD / 128, kB * kH);
    gemm_nt_bf16<<<gproj, 256, 0, stream>>>(
        xb, Wqb, Qb, (long long)kS * kD, (long long)kD * kD, (long long)kS * kD,
        kH, kS, kD, kD, kD, 0);
    gemm_nt_bf16<<<gproj, 256, 0, stream>>>(
        xb, Wkb, Kbf, (long long)kS * kD, (long long)kD * kD, (long long)kS * kD,
        kH, kS, kD, kD, kD, 0);
    gemm_nt_bf16<<<gproj, 256, 0, stream>>>(
        xb, Wvb, Vbf, (long long)kS * kD, (long long)kD * kD, (long long)kS * kD,
        kH, kS, kD, kD, kD, 0);

    // Fused flash attention -> cat [B,S,H*D] bf16
    const size_t smem = (size_t)(32 * LDK + 512 * LDV) * 2
                      + (size_t)64 * 32 * 4
                      + (size_t)64 * LDP * 2
                      + (size_t)64 * 3 * 4
                      + (size_t)(64 + 32) * 4;
    (void)hipFuncSetAttribute((const void*)flash_attn_bf16,
                              hipFuncAttributeMaxDynamicSharedMemorySize,
                              (int)smem);
    dim3 gattn(kB * kH, kS / 64);
    flash_attn_bf16<<<gattn, 256, smem, stream>>>(Qb, Kbf, Vbf, mask, cat);

    // Output projection: out[b] (SxD fp32) = cat[b] (SxHD) * Wp^T
    dim3 gproj2(kS / 128, kD / 128, kB);
    gemm_nt_bf16<<<gproj2, 256, 0, stream>>>(
        cat, Wpb, out, (long long)kS * kHD, 0, (long long)kS * kD,
        1, kS, kD, kHD, kD, 1);
}
